// PhysicsInvariantAttention_78503412236720
// MI455X (gfx1250) — matrix operand order repaired and checked by reading, compiled, not run on hardware
//
#include <hip/hip_runtime.h>

// PhysicsInvariantAttention for MI455X (gfx1250, wave32, WMMA).
// N=1024, C=128, H=4, HD=32, HID=64.
//
// Algebraic refactor: never materialize geom[N,N,C].
//   geom_attn[h,i,j] = sum_m relu((P[i,m]+b1[m]) - P[j,m]) * A[i,h,m] + q[i,h].b2
// with P = pos@W1, A = q-precontracted W2.
// The relu contraction runs on v_wmma_f32_16x16x32_f16 via a block-diagonal
// K=1024 embedding (B' built from two per-lane registers + cndmask).
// Flash softmax in transposed orientation (softmax dim = WMMA M rows ->
// in-lane reduce + one half-wave shuffle). All relu arithmetic is packed f16.

#define NN     1024
#define CC     128
#define HH     4
#define HD     32
#define HID    64
#define NSPLIT 4
#define RS32   0.17677669529663687f  // 1/sqrt(32)

typedef _Float16 half_t;
typedef __attribute__((ext_vector_type(16))) _Float16 v16h;
typedef __attribute__((ext_vector_type(8)))  _Float16 v8h;
typedef __attribute__((ext_vector_type(8)))  float    v8f;
typedef __attribute__((ext_vector_type(4)))  float    v4f;

static __device__ inline v8f wmma_f16(v16h a, v16h b, v8f c) {
  // D = A(16x32 f16) x B(32x16 f16) + C(16x16 f32)
  return __builtin_amdgcn_wmma_f32_16x16x32_f16(false, a, false, b, (short)0, c, false, false);
}

// ---------------------------------------------------------------------------
// Prep: q,k,v projections, P = pos@W1 (f16, raw and b1-folded), A = q∘W2,
// cq = q.b2, packed pos. q is stored pre-scaled by 1/sqrt(HD) so the qk and
// geom products share one WMMA accumulator chain. v is stored transposed so
// the PV WMMA A-fragment loads are contiguous b128s.
// ---------------------------------------------------------------------------
__global__ __launch_bounds__(CC)
void prep_kernel(const float* __restrict__ x,  const float* __restrict__ y,
                 const float* __restrict__ pos,
                 const float* __restrict__ Wq, const float* __restrict__ bq,
                 const float* __restrict__ Wk, const float* __restrict__ bk,
                 const float* __restrict__ Wv, const float* __restrict__ bv,
                 const float* __restrict__ W1, const float* __restrict__ b1,
                 const float* __restrict__ W2, const float* __restrict__ b2,
                 half_t* __restrict__ qh, half_t* __restrict__ kh,
                 half_t* __restrict__ vTh, half_t* __restrict__ Ah,
                 half_t* __restrict__ Ph, half_t* __restrict__ Pbh,
                 float* __restrict__ cq, float* __restrict__ pd) {
  __shared__ float xs[CC], ys[CC], qs[CC], ps[4];
  const int n = blockIdx.x, t = threadIdx.x;
  xs[t] = x[n*CC + t];
  ys[t] = y[n*CC + t];
  if (t < 3) ps[t] = pos[n*3 + t];
  __syncthreads();

  float q = bq[t], k = bk[t], v = bv[t];
  for (int kk = 0; kk < CC; ++kk) {
    q += xs[kk] * Wq[kk*CC + t];
    k += ys[kk] * Wk[kk*CC + t];
    v += ys[kk] * Wv[kk*CC + t];
  }
  qs[t] = q;
  qh[n*CC + t]  = (half_t)(q * RS32);
  kh[n*CC + t]  = (half_t)k;
  vTh[t*NN + n] = (half_t)v;

  if (t < HID) {
    float p = 0.f;
    for (int d = 0; d < 3; ++d) p += ps[d] * W1[d*HID + t];
    Ph[n*HID + t]  = (half_t)p;            // j-side rows
    Pbh[n*HID + t] = (half_t)(p + b1[t]);  // i-side rows, b1 folded at f32
  }
  if (t == 0) {
    pd[n*4+0] = ps[0]; pd[n*4+1] = ps[1]; pd[n*4+2] = ps[2];
    pd[n*4+3] = ps[0]*ps[0] + ps[1]*ps[1] + ps[2]*ps[2];
  }
  __syncthreads();

  for (int r = 0; r < 2; ++r) {
    const int idx = r*CC + t;
    const int h = idx >> 6, m = idx & 63;
    float a = 0.f;
    for (int d = 0; d < HD; ++d) a += qs[h*HD + d] * W2[m*CC + h*HD + d];
    Ah[n*HH*HID + idx] = (half_t)a;
  }
  if (t < HH) {
    float c = 0.f;
    for (int d = 0; d < HD; ++d) c += qs[t*HD + d] * b2[t*HD + d];
    cq[n*HH + t] = c;
  }
}

// ---------------------------------------------------------------------------
// Flash attention. One wave (32 lanes) per workgroup.
// Grid: 64 i-tiles x 4 heads x NSPLIT j-partitions.
// Score tiles are transposed: D[row=j, col=i].
// ---------------------------------------------------------------------------
__global__ __launch_bounds__(32)
__attribute__((amdgpu_waves_per_eu(4)))
void attn_kernel(const half_t* __restrict__ qh, const half_t* __restrict__ kh,
                 const half_t* __restrict__ vTh, const half_t* __restrict__ Ah,
                 const half_t* __restrict__ Ph, const half_t* __restrict__ Pbh,
                 const float* __restrict__ cq, const float* __restrict__ pd,
                 float* __restrict__ Op, float* __restrict__ ml) {
  const int b     = blockIdx.x;
  const int sp    = b & (NSPLIT - 1);
  const int h     = (b >> 2) & (HH - 1);
  const int itile = b >> 4;
  const int i0    = itile * 16;
  const int lane  = threadIdx.x;
  const int li    = lane & 15;   // column index (N dim of WMMA)
  const int hi    = lane >> 4;   // upper/lower half-wave
  const int irow  = i0 + li;

  __shared__ __align__(16) half_t Pibh[16 * HID];  // (P[i]+b1) f16, i-tile
  __shared__ __align__(16) float  pdj[32 * 4];     // packed pos for j-step

  // Stage the i-tile's 16x64 f16 rows (contiguous copy, coalesced v8h).
  {
    const v8h* src = (const v8h*)(Pbh + i0*HID);
    v8h* dst = (v8h*)Pibh;
    for (int c = lane; c < (16*HID)/8; c += 32) dst[c] = src[c];
  }
  __syncthreads();

  // q B-fragment (K=d: e + 16*hi), constant over the whole WG.
  const v16h qfrag = *(const v16h*)(qh + irow*CC + h*HD + 16*hi);
  // Block-diagonal B' source: this lane's A[i=li, h, :] split into two v16h.
  const half_t* ab = Ah + (irow*HH + h)*HID + 16*hi;
  const v16h B0 = *(const v16h*)(ab);        // m = 16*hi + 0..15
  const v16h B1 = *(const v16h*)(ab + 32);   // m = 32 + 16*hi + 0..15
  const v16h hz = {};
  const v8h  z8 = {};

  const float cqi = cq[irow*HH + h];
  const float pix = pd[irow*4+0], piy = pd[irow*4+1], piz = pd[irow*4+2];

  float mstate = -__builtin_inff();
  float lstate = 0.f;
  v8f acc0 = {};   // O^T rows d=0..15
  v8f acc1 = {};   // O^T rows d=16..31

  for (int js = 0; js < 256/32; ++js) {
    const int j0 = sp*256 + js*32;
    ((v4f*)pdj)[lane] = ((const v4f*)pd)[j0 + lane];
    __syncthreads();

    float sc[16];
    #pragma unroll
    for (int t = 0; t < 2; ++t) {
      const int jb   = j0 + 16*t;
      const int jrow = jb + li;
      // k A-fragment: row j=jrow, K pattern 0..15 + 16*hi
      const half_t* kb = kh + jrow*CC + h*HD + 16*hi;
      const v8h k0 = *(const v8h*)(kb);
      const v8h k1 = *(const v8h*)(kb + 8);
      const v16h kfrag = __builtin_shufflevector(k0, k1,
          0,1,2,3,4,5,6,7,8,9,10,11,12,13,14,15);
      v8f zf = {};
      v8f S = wmma_f16(kfrag, qfrag, zf);       // S^T = (k q^T)/sqrt(HD)

      // This lane's P row (f16): the 32 halves its fragment pattern touches:
      // row offsets {0..15, 32..47} + 16*hi.
      const half_t* pjp = Ph + jrow*HID + 16*hi;
      v8h Pj[4];
      #pragma unroll
      for (int g = 0; g < 4; ++g) Pj[g] = *(const v8h*)(pjp + 32*(g >> 1) + 8*(g & 1));

      // geom chain: K = (i', m); 16 iterations x 2 chunks. Pure packed-f16:
      // per chunk 2 LDS b128 + 2 pk_sub + 2 pk_max + masked B' + 1 WMMA.
      #pragma unroll 4
      for (int c2 = 0; c2 < 16; ++c2) {
        const half_t* pb = Pibh + c2*HID + 16*hi;
        const v16h bsel0 = (li == c2) ? B0 : hz;
        const v16h bsel1 = (li == c2) ? B1 : hz;
        #pragma unroll
        for (int par = 0; par < 2; ++par) {        // chunk c = 2*c2 + par
          const int mb = par * 32;
          const v8h a0 = *(const v8h*)(pb + mb);
          const v8h a1 = *(const v8h*)(pb + mb + 8);
          const v8h r0 = __builtin_elementwise_max(a0 - Pj[2*par],     z8);
          const v8h r1 = __builtin_elementwise_max(a1 - Pj[2*par + 1], z8);
          const v16h af = __builtin_shufflevector(r0, r1,
              0,1,2,3,4,5,6,7,8,9,10,11,12,13,14,15);
          S = wmma_f16(af, par ? bsel1 : bsel0, S);
        }
      }

      // dist (exact fp32) + q.b2 constant
      #pragma unroll
      for (int r = 0; r < 8; ++r) {
        const float* pj = pdj + (16*t + r + 8*hi)*4;
        const float dx = pix - pj[0], dy = piy - pj[1], dz = piz - pj[2];
        const float d2 = dx*dx + dy*dy + dz*dz;
        sc[8*t + r] = S[r] + cqi - sqrtf(d2);
      }
    }

    // ---- online softmax over these 32 j's (column = li) ----
    float tmax = sc[0];
    #pragma unroll
    for (int e = 1; e < 16; ++e) tmax = fmaxf(tmax, sc[e]);
    tmax = fmaxf(tmax, __shfl_xor(tmax, 16, 32));
    const float mnew = fmaxf(mstate, tmax);
    const float corr = __expf(mstate - mnew);
    float p[16]; float psum = 0.f;
    #pragma unroll
    for (int e = 0; e < 16; ++e) { p[e] = __expf(sc[e] - mnew); psum += p[e]; }
    psum += __shfl_xor(psum, 16, 32);
    lstate = lstate * corr + psum;
    mstate = mnew;
    acc0 *= corr; acc1 *= corr;

    // ---- build P^T B-fragment (K=j: e + 16*hi) via half-wave swap ----
    v16h pfrag;
    #pragma unroll
    for (int e = 0; e < 8; ++e) {
      const float o0 = __shfl_xor(p[e],     16, 32);
      const float o1 = __shfl_xor(p[8 + e], 16, 32);
      const float lo = hi ? o1       : p[e];
      const float hv = hi ? p[8 + e] : o0;
      pfrag[e]     = (_Float16)lo;
      pfrag[8 + e] = (_Float16)hv;
    }

    // ---- V^T A-fragments (rows d, K=j) and O^T accumulation ----
    const half_t* vb0 = vTh + (h*HD + li)*NN + j0 + 16*hi;
    const v8h a0l = *(const v8h*)(vb0);
    const v8h a0h = *(const v8h*)(vb0 + 8);
    const v16h vf0 = __builtin_shufflevector(a0l, a0h,
        0,1,2,3,4,5,6,7,8,9,10,11,12,13,14,15);
    const half_t* vb1 = vTh + (h*HD + 16 + li)*NN + j0 + 16*hi;
    const v8h a1l = *(const v8h*)(vb1);
    const v8h a1h = *(const v8h*)(vb1 + 8);
    const v16h vf1 = __builtin_shufflevector(a1l, a1h,
        0,1,2,3,4,5,6,7,8,9,10,11,12,13,14,15);
    acc0 = wmma_f16(vf0, pfrag, acc0);
    acc1 = wmma_f16(vf1, pfrag, acc1);
    __syncthreads();
  }

  // ---- store unnormalized partials + (m, l) ----
  float* op = Op + ((itile*HH + h)*NSPLIT + sp)*512;
  #pragma unroll
  for (int r = 0; r < 8; ++r) {
    op[(r + 8*hi)*16 + li]        = acc0[r];
    op[(r + 8*hi + 16)*16 + li]   = acc1[r];
  }
  float* mlp = ml + ((itile*HH + h)*NSPLIT + sp)*32;
  if (hi == 0) { mlp[li] = mstate; mlp[16 + li] = lstate; }
}

// ---------------------------------------------------------------------------
// Combine NSPLIT partial flash results per (i-tile, head).
// ---------------------------------------------------------------------------
__global__ __launch_bounds__(512)
void combine_kernel(const float* __restrict__ Op, const float* __restrict__ ml,
                    float* __restrict__ out) {
  const int b = blockIdx.x;            // itile*H + h
  const int itile = b >> 2, h = b & 3;
  const int t  = threadIdx.x;
  const int d  = t >> 4, il = t & 15;
  const int base = b * NSPLIT;
  float M = -__builtin_inff();
  #pragma unroll
  for (int s = 0; s < NSPLIT; ++s) M = fmaxf(M, ml[(base + s)*32 + il]);
  float L = 0.f, o = 0.f;
  #pragma unroll
  for (int s = 0; s < NSPLIT; ++s) {
    const float w = __expf(ml[(base + s)*32 + il] - M);
    L += w * ml[(base + s)*32 + 16 + il];
    o += w * Op[(base + s)*512 + d*16 + il];
  }
  out[(itile*16 + il)*CC + h*HD + d] = o / L;
}

// ---------------------------------------------------------------------------
extern "C" void kernel_launch(void* const* d_in, const int* in_sizes, int n_in,
                              void* d_out, int out_size, void* d_ws, size_t ws_size,
                              hipStream_t stream) {
  const float* x   = (const float*)d_in[0];
  const float* y   = (const float*)d_in[1];
  const float* pos = (const float*)d_in[2];
  const float* Wq  = (const float*)d_in[3];
  const float* bq  = (const float*)d_in[4];
  const float* Wk  = (const float*)d_in[5];
  const float* bk  = (const float*)d_in[6];
  const float* Wv  = (const float*)d_in[7];
  const float* bv  = (const float*)d_in[8];
  const float* W1  = (const float*)d_in[9];
  const float* b1  = (const float*)d_in[10];
  const float* W2  = (const float*)d_in[11];
  const float* b2  = (const float*)d_in[12];
  float* out = (float*)d_out;

  // workspace layout (~3.9 MB)
  half_t* qh  = (half_t*)d_ws;            // N*C f16 (pre-scaled by 1/sqrt(HD))
  half_t* khp = qh  + NN*CC;              // N*C f16
  half_t* vTh = khp + NN*CC;              // C*N f16 (transposed v)
  half_t* Ahp = vTh + NN*CC;              // N*H*HID f16
  half_t* Ph  = Ahp + NN*HH*HID;          // N*HID f16 (pos@W1)
  half_t* Pbh = Ph  + NN*HID;             // N*HID f16 (pos@W1 + b1)
  float*  cqp = (float*)(Pbh + NN*HID);   // N*H f32
  float*  pdp = cqp + NN*HH;              // N*4 f32 (pos.xyz, |pos|^2)
  float*  Opp = pdp + NN*4;               // 64*H*NSPLIT*512 f32 partial O^T
  float*  mlp = Opp + 64*HH*NSPLIT*512;   // 64*H*NSPLIT*32 f32 (m, l)

  prep_kernel<<<NN, CC, 0, stream>>>(x, y, pos, Wq, bq, Wk, bk, Wv, bv,
                                     W1, b1, W2, b2,
                                     qh, khp, vTh, Ahp, Ph, Pbh, cqp, pdp);
  attn_kernel<<<64*HH*NSPLIT, 32, 0, stream>>>(qh, khp, vTh, Ahp, Ph, Pbh,
                                               cqp, pdp, Opp, mlp);
  combine_kernel<<<64*HH, 512, 0, stream>>>(Opp, mlp, out);
}
